// oracle_GRU_72206990180495
// MI455X (gfx1250) — matrix operand order repaired and checked by reading, not compiled, not run on hardware
//
#include <hip/hip_runtime.h>
#include <cstdint>
#include <cstddef>

// ---------------- model dims ----------------
#define BB 512
#define LL 128
#define FF 24
#define DD 8
#define HH 256
#define H3 768
#define NHEAD 4
#define HD 64
#define BL (BB*LL)
#define OUTW (3*LL + 3*FF)   // 456

// ---------------- types ----------------
typedef __bf16 bf16_t;
typedef bf16_t v16bf __attribute__((ext_vector_type(16)));
typedef float  v8f   __attribute__((ext_vector_type(8)));
typedef unsigned int v4u __attribute__((ext_vector_type(4)));
typedef int v8i __attribute__((ext_vector_type(8)));
typedef int v4i __attribute__((ext_vector_type(4)));

// ---- Tensor Data Mover availability (device pass only; arity per toolchain) ----
#ifndef ATH_TDM
#if defined(__HIP_DEVICE_COMPILE__) && __has_builtin(__builtin_amdgcn_tensor_load_to_lds)
#define ATH_TDM 1
#else
#define ATH_TDM 0
#endif
#endif

__device__ inline bf16_t f2bf(float f) {
  unsigned u = __builtin_bit_cast(unsigned, f);
  u += 0x7FFFu + ((u >> 16) & 1u);                 // round-to-nearest-even
  unsigned short s = (unsigned short)(u >> 16);
  return __builtin_bit_cast(bf16_t, s);
}
__device__ inline bf16_t bf_zero() {
  unsigned short s = 0; return __builtin_bit_cast(bf16_t, s);
}

__device__ inline float wred_sum(float v) {
  #pragma unroll
  for (int o = 16; o > 0; o >>= 1) v += __shfl_xor(v, o, 32);
  return v;
}
__device__ inline float wred_max(float v) {
  #pragma unroll
  for (int o = 16; o > 0; o >>= 1) v = fmaxf(v, __shfl_xor(v, o, 32));
  return v;
}

#if ATH_TDM
// 2D tile load via Tensor Data Mover: rows x row_elems (bf16) from global
// (row stride = row_stride_elems) into contiguous LDS at lds_dst.
// D# layout per CDNA5 ISA 8.3/8.4: group0 = {count=1, lds_addr, gaddr, type=2},
// group1 = {data_size=2B, tensor_dim0/1, tile_dim0/1, dim0_stride}.
__device__ inline void tdm_load_2d(void* lds_dst, const void* gsrc,
                                   unsigned row_elems, unsigned rows,
                                   unsigned row_stride_elems) {
  unsigned long long ga = (unsigned long long)gsrc;
  unsigned lds = (unsigned)(unsigned long long)lds_dst;   // LDS addr = low 32 bits
  v4u g0 = { 1u,                                          // count=1, user mode
             lds,
             (unsigned)ga,
             (unsigned)((ga >> 32) & 0x01FFFFFFu) | (2u << 30) };  // type=2 (image)
  v8i g1 = { (int)(1u << 16),                             // data_size=1 -> 2 bytes
             (int)((row_elems & 0xFFFFu) << 16),          // tensor_dim0 lo16
             (int)(((row_elems >> 16) & 0xFFFFu) | ((rows & 0xFFFFu) << 16)),
             (int)(((rows >> 16) & 0xFFFFu) | ((row_elems & 0xFFFFu) << 16)), // tile_dim0
             (int)(rows & 0xFFFFu),                       // tile_dim1 (tile_dim2=0)
             (int)row_stride_elems,                       // tensor_dim0_stride lo32
             0, 0 };
  v4i gz = { 0, 0, 0, 0 };
#if __clang_major__ >= 23
  v8i gz8 = { 0, 0, 0, 0, 0, 0, 0, 0 };
  __builtin_amdgcn_tensor_load_to_lds(g0, g1, gz, gz, gz8, 0);
#else
  __builtin_amdgcn_tensor_load_to_lds(g0, g1, gz, gz, 0);
#endif
}
#endif

// ---- WMMA fragment loaders (16x16x32 bf16, wave32; this code's contiguous k order, same for A and B) ----
// A (16x32, 16-bit): lanes 0-15 -> M=lane, K = 0..15
//                    lanes 16-31 -> M=lane-16, K = 16..31
__device__ inline v16bf load_a_frag(const bf16_t* A, int lda, int lane, int k0) {
  int m = lane & 15, half = lane >> 4;
  const bf16_t* p = A + (size_t)m * lda + k0 + half * 16;
  v16bf a;
  #pragma unroll
  for (int e = 0; e < 16; ++e) a[e] = p[e];
  return a;
}
// B (32x16): lanes 0-15 -> N=lane, K=0..15 ; lanes 16-31 -> N=lane-16, K=16..31.
// W stored row-major [N][K] (original torch weight layout) -> contiguous per lane.
__device__ inline v16bf load_b_frag(const bf16_t* W, int ldk, int lane, int n0, int k0) {
  int n = lane & 15, half = lane >> 4;
  const bf16_t* p = W + (size_t)(n0 + n) * ldk + k0 + half * 16;
  v16bf b;
  #pragma unroll
  for (int e = 0; e < 16; ++e) b[e] = p[e];
  return b;
}
__device__ inline v8f wmma_bf(v16bf a, v16bf b, v8f c) {
  return __builtin_amdgcn_wmma_f32_16x16x32_bf16(false, a, false, b, (short)0, c, false, false);
}

// ---------------- fp32 -> bf16 convert ----------------
__global__ void k_cvt(const float* __restrict__ src, bf16_t* __restrict__ dst, int n) {
  int i = blockIdx.x * blockDim.x + threadIdx.x;
  if (i < n) dst[i] = f2bf(src[i]);
}

// ---------------- generic 32-row-strip GEMM + fused epilogue ----------------
// C[32 x N] = act( LN( A[32 x K] @ W^T + bias + res ) )   (A packed [M][K])
// mode: 0 = bias only, 1 = bias+res+LN, 2 = bias+LN+ELU
template<int N, int K>
__global__ void k_rowstrip(const bf16_t* __restrict__ A,
                           const bf16_t* __restrict__ W,
                           const float* __restrict__ bias,
                           const float* __restrict__ res, int ldres,
                           const float* __restrict__ gamma, const float* __restrict__ beta,
                           int mode,
                           float* __restrict__ outf, int ldof,
                           bf16_t* __restrict__ outb, int ldob) {
  __shared__ bf16_t As[32 * K];
  __shared__ float  Cs[32 * N];
  const int m0 = blockIdx.x * 32;
  const int tid = threadIdx.x, lane = tid & 31, wave = tid >> 5;
  const bf16_t* Ablk = A + (size_t)m0 * K;

#if ATH_TDM
  if (wave == 0) {
    tdm_load_2d(As, Ablk, K, 32, K);
    __builtin_amdgcn_s_wait_tensorcnt(0);
  }
#else
  for (int idx = tid; idx < 32 * K; idx += 256) As[idx] = Ablk[idx];
#endif
  __syncthreads();

  // 2 M-tiles x (N/16) N-tiles, register-blocked 2 N-tiles per wave pass
  constexpr int NP = N / 32;
  for (int tp = wave; tp < 2 * NP; tp += 8) {
    const int mt = tp & 1, n0 = (tp >> 1) * 32;
    __builtin_prefetch(W + (size_t)(n0 + (lane & 15)) * K, 0, 1);
    v8f acc0 = {}, acc1 = {};
    #pragma unroll 4
    for (int k0 = 0; k0 < K; k0 += 32) {
      v16bf a  = load_a_frag(As + mt * 16 * K, K, lane, k0);
      v16bf b0 = load_b_frag(W, K, lane, n0, k0);
      v16bf b1 = load_b_frag(W, K, lane, n0 + 16, k0);
      acc0 = wmma_bf(a, b0, acc0);
      acc1 = wmma_bf(a, b1, acc1);
    }
    #pragma unroll
    for (int i = 0; i < 8; ++i) {
      const int m = mt * 16 + (lane >> 4) * 8 + i, n = n0 + (lane & 15);
      Cs[m * N + n]      = acc0[i];
      Cs[m * N + n + 16] = acc1[i];
    }
  }
  __syncthreads();

  for (int r = wave * 4; r < wave * 4 + 4; ++r) {
    const int m = m0 + r;
    for (int n = lane; n < N; n += 32) {
      float v = Cs[r * N + n];
      if (bias) v += bias[n];
      if (res)  v += res[(size_t)m * ldres + n];
      Cs[r * N + n] = v;
    }
    if (mode != 0) {
      float s = 0.f;
      for (int n = lane; n < N; n += 32) s += Cs[r * N + n];
      const float mean = wred_sum(s) / (float)N;
      float vs = 0.f;
      for (int n = lane; n < N; n += 32) { float d = Cs[r * N + n] - mean; vs += d * d; }
      const float inv = rsqrtf(wred_sum(vs) / (float)N + 1e-5f);
      for (int n = lane; n < N; n += 32) {
        float d = (Cs[r * N + n] - mean) * inv * gamma[n] + beta[n];
        if (mode == 2) d = d > 0.f ? d : (__expf(d) - 1.f);   // ELU
        Cs[r * N + n] = d;
      }
    }
    for (int n = lane; n < N; n += 32) {
      float v = Cs[r * N + n];
      if (outf) outf[(size_t)m * ldof + n] = v;
      if (outb) outb[(size_t)m * ldob + n] = f2bf(v);
    }
  }
}

// ---------------- fused GRU step (encoder & decoder) ----------------
// One wave per 16(batch) x 16(hidden) output tile; 3 gate GEMMs K=256 via WMMA.
__global__ void k_gru_step(const bf16_t* __restrict__ hprev_b, size_t hpb_stride,
                           const float* __restrict__ hprev_f, size_t hpf_stride,
                           const float* __restrict__ x, size_t x_stride, int Dx,
                           const float* __restrict__ Wi,      // [768][Dx] fp32
                           const bf16_t* __restrict__ Wh,     // [768][256] bf16
                           const float* __restrict__ bi, const float* __restrict__ bh,
                           float* __restrict__ hout_f, size_t hof_stride,
                           bf16_t* __restrict__ hout_b, size_t hob_stride) {
  __shared__ bf16_t As[16 * 256];
  const int b0 = blockIdx.x * 16, n0t = blockIdx.y * 16;
  const int lane = threadIdx.x;

  if (hprev_b) {
#if ATH_TDM
    tdm_load_2d(As, hprev_b + (size_t)b0 * hpb_stride, 256, 16, (unsigned)hpb_stride);
    __builtin_amdgcn_s_wait_tensorcnt(0);
#else
    for (int idx = lane; idx < 16 * 256; idx += 32) {
      int r = idx >> 8, c = idx & 255;
      As[idx] = hprev_b[(size_t)(b0 + r) * hpb_stride + c];
    }
#endif
  } else {
    for (int idx = lane; idx < 16 * 256; idx += 32) As[idx] = bf_zero();
  }
  __syncthreads();

  v8f accr = {}, accz = {}, accn = {};
  #pragma unroll
  for (int k0 = 0; k0 < 256; k0 += 32) {
    v16bf a = load_a_frag(As, 256, lane, k0);
    accr = wmma_bf(a, load_b_frag(Wh, 256, lane,       n0t, k0), accr);
    accz = wmma_bf(a, load_b_frag(Wh, 256, lane, 256 + n0t, k0), accz);
    accn = wmma_bf(a, load_b_frag(Wh, 256, lane, 512 + n0t, k0), accn);
  }

  #pragma unroll
  for (int i = 0; i < 8; ++i) {
    int m = b0 + ((lane >> 4) * 8 + i);
    int n = n0t + (lane & 15);
    float gir = bi[n], giz = bi[256 + n], gin = bi[512 + n];
    for (int d = 0; d < Dx; ++d) {
      float xv = x[(size_t)m * x_stride + d];
      gir += xv * Wi[(size_t)n * Dx + d];
      giz += xv * Wi[(size_t)(256 + n) * Dx + d];
      gin += xv * Wi[(size_t)(512 + n) * Dx + d];
    }
    float ghr = accr[i] + bh[n];
    float ghz = accz[i] + bh[256 + n];
    float ghn = accn[i] + bh[512 + n];
    float r = 1.f / (1.f + __expf(-(gir + ghr)));
    float z = 1.f / (1.f + __expf(-(giz + ghz)));
    float nn = tanhf(gin + r * ghn);
    float hp = hprev_f ? hprev_f[(size_t)m * hpf_stride + n] : 0.f;
    float h = (1.f - z) * nn + z * hp;
    hout_f[(size_t)m * hof_stride + n] = h;
    hout_b[(size_t)m * hob_stride + n] = f2bf(h);
  }
}

// ---------------- fused per-(batch,head) attention ----------------
__global__ void k_attn(const bf16_t* __restrict__ encb,   // [B][L][H] bf16
                       const bf16_t* __restrict__ Winp,   // [768][256] bf16
                       const float* __restrict__ binp,    // [768]
                       bf16_t* __restrict__ mh) {         // [B*L][H] bf16 (pre out-proj)
  __shared__ bf16_t q_s[128 * 64];
  __shared__ bf16_t k_s[128 * 64];
  __shared__ bf16_t vT_s[64 * 128];
  __shared__ float  S_s[128 * 128];
  __shared__ bf16_t P_s[128 * 128];

  const int b = blockIdx.x, head = blockIdx.y;
  const int tid = threadIdx.x, lane = tid & 31, wave = tid >> 5;
  const bf16_t* A = encb + (size_t)b * LL * HH;

  // q/k/v projections: 3 mats * 8 Mtiles * 4 Ntiles = 96 tile-GEMMs, K=256
  for (int t = wave; t < 96; t += 8) {
    int p = t / 32, rem = t - p * 32, mt = rem >> 2, nt = rem & 3;
    int nbase = p * 256 + head * 64 + nt * 16;
    v8f acc = {};
    #pragma unroll 4
    for (int k0 = 0; k0 < 256; k0 += 32) {
      v16bf a = load_a_frag(A + (size_t)mt * 16 * HH, HH, lane, k0);
      v16bf bb = load_b_frag(Winp, HH, lane, nbase, k0);
      acc = wmma_bf(a, bb, acc);
    }
    #pragma unroll
    for (int i = 0; i < 8; ++i) {
      int m = mt * 16 + ((lane >> 4) * 8 + i);
      int nl = nt * 16 + (lane & 15);
      float v = acc[i] + binp[nbase + (lane & 15)];
      bf16_t bv = f2bf(v);
      if (p == 0)      q_s[m * 64 + nl] = bv;
      else if (p == 1) k_s[m * 64 + nl] = bv;
      else             vT_s[nl * 128 + m] = bv;   // transposed store
    }
  }
  __syncthreads();

  // S = (q @ k^T) / 8 : 8x8 tiles, K=64
  for (int t = wave; t < 64; t += 8) {
    int mt = t >> 3, nt = t & 7;
    v8f acc = {};
    #pragma unroll
    for (int k0 = 0; k0 < 64; k0 += 32)
      acc = wmma_bf(load_a_frag(q_s + (size_t)mt * 16 * 64, 64, lane, k0),
                    load_b_frag(k_s, 64, lane, nt * 16, k0), acc);
    #pragma unroll
    for (int i = 0; i < 8; ++i) {
      int m = mt * 16 + ((lane >> 4) * 8 + i), n = nt * 16 + (lane & 15);
      S_s[m * 128 + n] = acc[i] * 0.125f;
    }
  }
  __syncthreads();

  // row softmax -> P (bf16)
  for (int r = wave * 16; r < wave * 16 + 16; ++r) {
    float mx = -1e30f;
    for (int c = lane; c < 128; c += 32) mx = fmaxf(mx, S_s[r * 128 + c]);
    mx = wred_max(mx);
    float s = 0.f;
    for (int c = lane; c < 128; c += 32) {
      float e = __expf(S_s[r * 128 + c] - mx);
      S_s[r * 128 + c] = e; s += e;
    }
    float inv = 1.f / wred_sum(s);
    for (int c = lane; c < 128; c += 32) P_s[r * 128 + c] = f2bf(S_s[r * 128 + c] * inv);
  }
  __syncthreads();

  // O = P @ v : 8x4 tiles, K=128  (v stored transposed -> standard B layout)
  for (int t = wave; t < 32; t += 8) {
    int mt = t >> 2, nt = t & 3;
    v8f acc = {};
    #pragma unroll
    for (int k0 = 0; k0 < 128; k0 += 32)
      acc = wmma_bf(load_a_frag(P_s + (size_t)mt * 16 * 128, 128, lane, k0),
                    load_b_frag(vT_s, 128, lane, nt * 16, k0), acc);
    #pragma unroll
    for (int i = 0; i < 8; ++i) {
      int l = mt * 16 + ((lane >> 4) * 8 + i);
      int d = nt * 16 + (lane & 15);
      mh[((size_t)b * LL + l) * HH + head * 64 + d] = f2bf(acc[i]);
    }
  }
}

// ---------------- small helper kernels ----------------
__global__ void k_mean(const float* __restrict__ encf, float* __restrict__ hf,
                       bf16_t* __restrict__ hb) {
  int b = blockIdx.x, h = threadIdx.x;
  float s = 0.f;
  for (int l = 0; l < LL; ++l) s += encf[((size_t)b * LL + l) * HH + h];
  s *= (1.f / (float)LL);
  hf[(size_t)b * HH + h] = s;
  hb[(size_t)b * HH + h] = f2bf(s);
}

__global__ void k_dec_w(const float* __restrict__ hdec, const float* __restrict__ gpfm,
                        const float* __restrict__ prev, int t, int first,
                        const float* __restrict__ Winf, const float* __restrict__ binf,
                        float* __restrict__ w, float* __restrict__ indec) {
  __shared__ float redm[4], reds[4];
  int b = blockIdx.x, j = threadIdx.x, lane = j & 31, wv = j >> 5;
  float ind = first ? gpfm[(size_t)b * FF + t] : prev[b];
  const float* Wr = Winf + (size_t)j * (HH + 1);
  const float* hr = hdec + (size_t)b * HH;
  float s = binf[j] + ind * Wr[HH];
  for (int k = 0; k < HH; ++k) s += hr[k] * Wr[k];
  float mx = wred_max(s);
  if (lane == 0) redm[wv] = mx;
  __syncthreads();
  mx = fmaxf(fmaxf(redm[0], redm[1]), fmaxf(redm[2], redm[3]));
  float e = __expf(s - mx);
  float ss = wred_sum(e);
  if (lane == 0) reds[wv] = ss;
  __syncthreads();
  ss = reds[0] + reds[1] + reds[2] + reds[3];
  w[(size_t)b * LL + j] = e / ss;
  if (j == 0) indec[b] = ind;
}

__global__ void k_dec_applied(const float* __restrict__ w, const float* __restrict__ encf,
                              const float* __restrict__ hdecf, bf16_t* __restrict__ combin) {
  __shared__ float ws[LL];
  int b = blockIdx.x, h = threadIdx.x;
  if (h < LL) ws[h] = w[(size_t)b * LL + h];
  __syncthreads();
  float s = 0.f;
  for (int l = 0; l < LL; ++l) s += ws[l] * encf[((size_t)b * LL + l) * HH + h];
  combin[(size_t)b * 512 + h]       = f2bf(s);
  combin[(size_t)b * 512 + 256 + h] = f2bf(hdecf[(size_t)b * HH + h]);
}

__global__ void k_dec_out(const float* __restrict__ z2f, const float* __restrict__ Woutc,
                          const float* __restrict__ boutc,
                          const float* __restrict__ gpm, const float* __restrict__ gpl,
                          const float* __restrict__ gpu,
                          int t, float* __restrict__ prev, float* __restrict__ out) {
  int b = blockIdx.x * blockDim.x + threadIdx.x;
  if (b >= BB) return;
  float s = boutc[0];
  for (int k = 0; k < 128; ++k) s += z2f[(size_t)b * 128 + k] * Woutc[k];
  float pm = gpm[(size_t)b * FF + t];
  prev[b] = s + pm;
  float* row = out + (size_t)b * OUTW;
  row[3 * LL + t]            = s + pm;
  row[3 * LL + FF + t]       = s + gpl[(size_t)b * FF + t];
  row[3 * LL + 2 * FF + t]   = s + gpu[(size_t)b * FF + t];
}

__global__ void k_past_out(const float* __restrict__ z2f, int row0, int nrows,
                           const float* __restrict__ Woutc, const float* __restrict__ boutc,
                           const float* __restrict__ gpm, const float* __restrict__ gpl,
                           const float* __restrict__ gpu, float* __restrict__ out) {
  int r = blockIdx.x * blockDim.x + threadIdx.x;
  if (r >= nrows) return;
  int g = row0 + r, b = g / LL, l = g - b * LL;
  float s = boutc[0];
  for (int k = 0; k < 128; ++k) s += z2f[(size_t)r * 128 + k] * Woutc[k];
  float* row = out + (size_t)b * OUTW;
  row[l]          = s + gpm[g];
  row[LL + l]     = s + gpl[g];
  row[2 * LL + l] = s + gpu[g];
}

// ---------------- host launcher ----------------
extern "C" void kernel_launch(void* const* d_in, const int* in_sizes, int n_in,
                              void* d_out, int out_size, void* d_ws, size_t ws_size,
                              hipStream_t stream) {
  (void)in_sizes; (void)n_in; (void)out_size; (void)ws_size;
  const float* x            = (const float*)d_in[0];
  const float* gp_past_mean = (const float*)d_in[1];
  const float* gp_past_low  = (const float*)d_in[2];
  const float* gp_past_up   = (const float*)d_in[3];
  const float* gp_fut_mean  = (const float*)d_in[4];
  const float* gp_fut_low   = (const float*)d_in[5];
  const float* gp_fut_up    = (const float*)d_in[6];
  const float* W_ih_enc = (const float*)d_in[7];
  const float* W_hh_enc = (const float*)d_in[8];
  const float* b_ih_enc = (const float*)d_in[9];
  const float* b_hh_enc = (const float*)d_in[10];
  const float* W_in_proj = (const float*)d_in[11];
  const float* b_in_proj = (const float*)d_in[12];
  const float* W_out_proj = (const float*)d_in[13];
  const float* b_out_proj = (const float*)d_in[14];
  const float* g_mh_ln = (const float*)d_in[15];
  const float* be_mh_ln = (const float*)d_in[16];
  const float* W_inf = (const float*)d_in[17];
  const float* b_inf = (const float*)d_in[18];
  const float* W_comb = (const float*)d_in[19];
  const float* b_comb = (const float*)d_in[20];
  const float* g_inf_ln = (const float*)d_in[21];
  const float* be_inf_ln = (const float*)d_in[22];
  const float* W_ih_dec = (const float*)d_in[23];
  const float* W_hh_dec = (const float*)d_in[24];
  const float* b_ih_dec = (const float*)d_in[25];
  const float* b_hh_dec = (const float*)d_in[26];
  const float* W_h0 = (const float*)d_in[27];
  const float* b_h0 = (const float*)d_in[28];
  const float* g_ln0 = (const float*)d_in[29];
  const float* be_ln0 = (const float*)d_in[30];
  const float* W_h1 = (const float*)d_in[31];
  const float* b_h1 = (const float*)d_in[32];
  const float* g_ln1 = (const float*)d_in[33];
  const float* be_ln1 = (const float*)d_in[34];
  const float* W_h2 = (const float*)d_in[35];
  const float* b_h2 = (const float*)d_in[36];
  const float* g_ln2 = (const float*)d_in[37];
  const float* be_ln2 = (const float*)d_in[38];
  const float* W_outc = (const float*)d_in[39];
  const float* b_outc = (const float*)d_in[40];
  float* out = (float*)d_out;

  // ---- workspace bump allocator ----
  char* ws = (char*)d_ws;
  size_t off = 0;
  auto alloc = [&](size_t bytes) -> char* {
    char* p = ws + off;
    off = (off + bytes + 255) & ~(size_t)255;
    return p;
  };
  bf16_t* Whh_b    = (bf16_t*)alloc(H3 * HH * 2);
  bf16_t* Winp_b   = (bf16_t*)alloc(H3 * HH * 2);
  bf16_t* Woutp_b  = (bf16_t*)alloc(HH * HH * 2);
  bf16_t* Wcomb_b  = (bf16_t*)alloc(HH * 512 * 2);
  bf16_t* Whhdec_b = (bf16_t*)alloc(H3 * HH * 2);
  bf16_t* Wh0_b    = (bf16_t*)alloc(512 * HH * 2);
  bf16_t* Wh1_b    = (bf16_t*)alloc(HH * 512 * 2);
  bf16_t* Wh2_b    = (bf16_t*)alloc(128 * HH * 2);
  float*  enc_f    = (float*)alloc((size_t)BL * HH * 4);
  bf16_t* enc_b    = (bf16_t*)alloc((size_t)BL * HH * 2);
  char*   big      = alloc((size_t)BL * HH * 2);      // 32MB reused: mh / past z-bufs
  bf16_t* mh_b     = (bf16_t*)big;
  // decoder scratch
  float*  hdec_f  = (float*)alloc((size_t)BB * HH * 4);
  bf16_t* hdec_b  = (bf16_t*)alloc((size_t)BB * HH * 2);
  float*  hmid_f  = (float*)alloc((size_t)BB * HH * 4);
  bf16_t* hmid_b  = (bf16_t*)alloc((size_t)BB * HH * 2);
  bf16_t* combin  = (bf16_t*)alloc((size_t)BB * 512 * 2);
  float*  wbuf    = (float*)alloc((size_t)BB * LL * 4);
  float*  indec   = (float*)alloc(BB * 4);
  float*  prevb   = (float*)alloc(BB * 4);
  bf16_t* z0d     = (bf16_t*)alloc((size_t)BB * 512 * 2);
  bf16_t* z1d     = (bf16_t*)alloc((size_t)BB * HH * 2);
  float*  z2d_f   = (float*)alloc((size_t)BB * 128 * 4);

  // ---- 1) weight conversion to bf16 ([N][K] row-major kept as-is) ----
  auto cvt = [&](const float* s, bf16_t* d, int n) {
    k_cvt<<<(n + 255) / 256, 256, 0, stream>>>(s, d, n);
  };
  cvt(W_hh_enc,  Whh_b,    H3 * HH);
  cvt(W_in_proj, Winp_b,   H3 * HH);
  cvt(W_out_proj,Woutp_b,  HH * HH);
  cvt(W_comb,    Wcomb_b,  HH * 512);
  cvt(W_hh_dec,  Whhdec_b, H3 * HH);
  cvt(W_h0,      Wh0_b,    512 * HH);
  cvt(W_h1,      Wh1_b,    HH * 512);
  cvt(W_h2,      Wh2_b,    128 * HH);

  // ---- 2) encoder GRU: 128 sequential WMMA steps, h written into enc ----
  for (int t = 0; t < LL; ++t) {
    const bf16_t* hpb = t ? enc_b + (size_t)(t - 1) * HH : (const bf16_t*)nullptr;
    const float*  hpf = t ? enc_f + (size_t)(t - 1) * HH : (const float*)nullptr;
    k_gru_step<<<dim3(BB / 16, HH / 16), 32, 0, stream>>>(
        hpb, (size_t)LL * HH, hpf, (size_t)LL * HH,
        x + (size_t)t * DD, (size_t)LL * DD, DD,
        W_ih_enc, Whh_b, b_ih_enc, b_hh_enc,
        enc_f + (size_t)t * HH, (size_t)LL * HH,
        enc_b + (size_t)t * HH, (size_t)LL * HH);
  }

  // ---- 3) fused attention (per batch x head) ----
  k_attn<<<dim3(BB, NHEAD), 256, 0, stream>>>(enc_b, Winp_b, b_in_proj, mh_b);

  // ---- 4) out-proj + residual + LayerNorm (in-place over enc) ----
  k_rowstrip<HH, HH><<<BL / 32, 256, 0, stream>>>(mh_b, Woutp_b, b_out_proj,
                                                  enc_f, HH, g_mh_ln, be_mh_ln, 1,
                                                  enc_f, HH, enc_b, HH);

  // ---- 5) h_dec0 = mean over L ----
  k_mean<<<BB, HH, 0, stream>>>(enc_f, hdec_f, hdec_b);

  // ---- 6) autoregressive decoder, 24 steps ----
  for (int t = 0; t < FF; ++t) {
    k_dec_w<<<BB, LL, 0, stream>>>(hdec_f, gp_fut_mean, prevb, t, t == 0,
                                   W_inf, b_inf, wbuf, indec);
    k_dec_applied<<<BB, HH, 0, stream>>>(wbuf, enc_f, hdec_f, combin);
    k_rowstrip<HH, 512><<<BB / 32, 256, 0, stream>>>(combin, Wcomb_b, b_comb,
                                                     hdec_f, HH, g_inf_ln, be_inf_ln, 1,
                                                     hmid_f, HH, hmid_b, HH);
    k_gru_step<<<dim3(BB / 16, HH / 16), 32, 0, stream>>>(
        hmid_b, (size_t)HH, hmid_f, (size_t)HH,
        indec, (size_t)1, 1,
        W_ih_dec, Whhdec_b, b_ih_dec, b_hh_dec,
        hdec_f, (size_t)HH, hdec_b, (size_t)HH);
    // MLP head on h_dec
    k_rowstrip<512, HH><<<BB / 32, 256, 0, stream>>>(hdec_b, Wh0_b, b_h0,
                                                     nullptr, 0, g_ln0, be_ln0, 2,
                                                     nullptr, 0, z0d, 512);
    k_rowstrip<HH, 512><<<BB / 32, 256, 0, stream>>>(z0d, Wh1_b, b_h1,
                                                     nullptr, 0, g_ln1, be_ln1, 2,
                                                     nullptr, 0, z1d, HH);
    k_rowstrip<128, HH><<<BB / 32, 256, 0, stream>>>(z1d, Wh2_b, b_h2,
                                                     nullptr, 0, g_ln2, be_ln2, 2,
                                                     z2d_f, 128, nullptr, 0);
    k_dec_out<<<(BB + 255) / 256, 256, 0, stream>>>(z2d_f, W_outc, b_outc,
                                                    gp_fut_mean, gp_fut_low, gp_fut_up,
                                                    t, prevb, out);
  }

  // ---- 7) past branch: MLP over all 65536 encoder states (chunked) ----
  const int CH = 8192;
  bf16_t* z0p   = (bf16_t*)big;                                  // 8 MB
  bf16_t* z1p   = (bf16_t*)(big + (size_t)CH * 512 * 2);         // 4 MB
  float*  z2p_f = (float*)(big + (size_t)CH * 512 * 2 + (size_t)CH * HH * 2);
  for (int c = 0; c < BL / CH; ++c) {
    const bf16_t* Ain = enc_b + (size_t)c * CH * HH;
    k_rowstrip<512, HH><<<CH / 32, 256, 0, stream>>>(Ain, Wh0_b, b_h0,
                                                     nullptr, 0, g_ln0, be_ln0, 2,
                                                     nullptr, 0, z0p, 512);
    k_rowstrip<HH, 512><<<CH / 32, 256, 0, stream>>>(z0p, Wh1_b, b_h1,
                                                     nullptr, 0, g_ln1, be_ln1, 2,
                                                     nullptr, 0, z1p, HH);
    k_rowstrip<128, HH><<<CH / 32, 256, 0, stream>>>(z1p, Wh2_b, b_h2,
                                                     nullptr, 0, g_ln2, be_ln2, 2,
                                                     z2p_f, 128, nullptr, 0);
    k_past_out<<<CH / 256, 256, 0, stream>>>(z2p_f, c * CH, CH, W_outc, b_outc,
                                             gp_past_mean, gp_past_low, gp_past_up, out);
  }
}
